// GETS_27393301414251
// MI455X (gfx1250) — matrix operand order repaired and checked by reading, not compiled, not run on hardware
//
#include <hip/hip_runtime.h>
#include <hip/hip_fp16.h>

typedef __attribute__((ext_vector_type(16))) _Float16 v16h;
typedef __attribute__((ext_vector_type(8)))  _Float16 v8h;
typedef __attribute__((ext_vector_type(8)))  float    v8f;

#define NODES 50000
#define EDGES 800000
#define KD    576   // 512 feats + 64 logits
#define H1    256
#define OUTD  64
#define NE    4

__device__ __forceinline__ float atomAddF(float* p, float v) {
  return __hip_atomic_fetch_add(p, v, __ATOMIC_RELAXED, __HIP_MEMORY_SCOPE_AGENT);
}

// ---------------- degree / norms ----------------
__global__ void k_degree(const int* __restrict__ src, const int* __restrict__ dst,
                         float* deg_out, float* deg_in) {
  int i = blockIdx.x * blockDim.x + threadIdx.x;
  if (i < EDGES) {
    atomAddF(&deg_out[src[i]], 1.0f);
    atomAddF(&deg_in[dst[i]], 1.0f);
  }
}
__global__ void k_norm(float* norm_s, float* norm_d) {
  int i = blockIdx.x * blockDim.x + threadIdx.x;
  if (i < NODES) {
    norm_s[i] = rsqrtf(fmaxf(norm_s[i], 1.0f));
    norm_d[i] = rsqrtf(fmaxf(norm_d[i], 1.0f));
  }
}

// ---------------- activation convert: Acat = f16(concat(features, logits)) ----------------
__global__ void k_acat(const float* __restrict__ feats, const float* __restrict__ logits,
                       _Float16* __restrict__ acat) {
  long long idx = (long long)blockIdx.x * blockDim.x + threadIdx.x;
  if (idx >= (long long)NODES * KD) return;
  int n = (int)(idx / KD), k = (int)(idx % KD);
  float v = (k < 512) ? feats[(size_t)n * 512 + k] : logits[(size_t)n * 64 + (k - 512)];
  acat[idx] = (_Float16)v;
}

// ---------------- fused gate weights: wg[k][e], biasg[e] ----------------
__global__ void k_prep_wg(const float* __restrict__ Wp_gate, const float* __restrict__ w_gate,
                          float* __restrict__ wg) {
  int idx = blockIdx.x * blockDim.x + threadIdx.x;
  if (idx >= KD * NE) return;
  int k = idx >> 2, e = idx & 3;
  float acc = 0.f;
  if (k < 512) {
    for (int j = 0; j < 256; ++j) acc += Wp_gate[(size_t)k * 256 + j] * w_gate[j * NE + e];
  } else {
    acc = w_gate[(256 + (k - 512)) * NE + e];
  }
  wg[idx] = acc;
}
__global__ void k_prep_biasg(const float* __restrict__ bp_gate, const float* __restrict__ w_gate,
                             float* __restrict__ biasg) {
  int e = threadIdx.x;
  if (e >= NE) return;
  float acc = 0.f;
  for (int j = 0; j < 256; ++j) acc += bp_gate[j] * w_gate[j * NE + e];
  biasg[e] = acc;
}

// ---------------- fused expert layer-1 weights ----------------
// bt1[e][c][k] : k<512 -> sum_j Wp_e[e][k][j]*W1[e][64+j][c] ; k>=512 -> W1[e][k-512][c]
__global__ void k_prep_bt1(const float* __restrict__ Wp_e, const float* __restrict__ W1,
                           _Float16* __restrict__ bt1) {
  int idx = blockIdx.x * blockDim.x + threadIdx.x;
  if (idx >= NE * H1 * KD) return;
  int e = idx / (H1 * KD);
  int rem = idx % (H1 * KD);
  int c = rem / KD, k = rem % KD;
  float acc;
  if (k < 512) {
    acc = 0.f;
    const float* wp = Wp_e + (size_t)e * 512 * 256 + (size_t)k * 256;
    const float* w1 = W1 + (size_t)e * 320 * 256 + (size_t)64 * 256 + c;
    for (int j = 0; j < 256; ++j) acc += wp[j] * w1[(size_t)j * 256];
  } else {
    acc = W1[(size_t)e * 320 * 256 + (size_t)(k - 512) * 256 + c];
  }
  bt1[idx] = (_Float16)acc;
}
__global__ void k_prep_b1f(const float* __restrict__ bp_e, const float* __restrict__ W1,
                           float* __restrict__ b1f) {
  int idx = blockIdx.x * blockDim.x + threadIdx.x;
  if (idx >= NE * H1) return;
  int e = idx / H1, c = idx % H1;
  float acc = 0.f;
  for (int j = 0; j < 256; ++j)
    acc += bp_e[(size_t)e * 256 + j] * W1[(size_t)e * 320 * 256 + (size_t)(64 + j) * 256 + c];
  b1f[idx] = acc;
}
// bt2[e][c][k] = W2[e][k][c]  (transpose + f16)
__global__ void k_prep_bt2(const float* __restrict__ W2, _Float16* __restrict__ bt2) {
  int idx = blockIdx.x * blockDim.x + threadIdx.x;
  if (idx >= NE * OUTD * H1) return;
  int e = idx / (OUTD * H1);
  int rem = idx % (OUTD * H1);
  int c = rem / H1, k = rem % H1;
  bt2[idx] = (_Float16)W2[(size_t)e * H1 * OUTD + (size_t)k * OUTD + c];
}

// ---------------- WMMA fragment loaders (ISA 16-bit layouts, wave32) ----------------
__device__ __forceinline__ v16h loadA16(const _Float16* __restrict__ A, int row, int lda,
                                        int k0, int lane) {
  int up = lane >> 4;
  const _Float16* p = A + (size_t)row * lda + k0 + up * 8;
  v8h lo = *(const v8h*)p;        // K = k0+up*8   .. +7
  v8h hi = *(const v8h*)(p + 16); // K = k0+16+up*8 .. +7
  v16h r;
#pragma unroll
  for (int i = 0; i < 8; ++i) { r[i] = lo[i]; r[i + 8] = hi[i]; }
  return r;
}
__device__ __forceinline__ v16h loadB16(const _Float16* __restrict__ Bt, int col, int ldb,
                                        int k0, int lane) {
  int up = lane >> 4;
  const _Float16* p = Bt + (size_t)col * ldb + k0 + up * 8; // K = k0+up*8 .. +7 ; k0+16+up*8 .. +7
  v8h lo = *(const v8h*)p;
  v8h hi = *(const v8h*)(p + 16);
  v16h r;
#pragma unroll
  for (int i = 0; i < 8; ++i) { r[i] = lo[i]; r[i + 8] = hi[i]; }
  return r;
}

#define WMMA_F16(a, b, c) \
  __builtin_amdgcn_wmma_f32_16x16x32_f16(false, (a), false, (b), (short)0, (c), false, false)

// ---------------- GEMM1: e1 = (Acat @ Bt1^T + b1f) * norm_s ; C [NODES,256] ----------------
__global__ void __launch_bounds__(256)
k_gemm1(const _Float16* __restrict__ A, const _Float16* __restrict__ Bt,
        const float* __restrict__ bias, const float* __restrict__ norm_s,
        float* __restrict__ C) {
  int lane = threadIdx.x & 31, wave = threadIdx.x >> 5;
  int hl = lane & 15, up = lane >> 4;
  int rowbase = blockIdx.x * 128 + wave * 16;
  int colbase = blockIdx.y * 64;
  int arow = rowbase + hl; if (arow >= NODES) arow = NODES - 1;
  v8f c0 = {}, c1 = {}, c2 = {}, c3 = {};
  for (int k0 = 0; k0 < KD; k0 += 32) {
    v16h a  = loadA16(A, arow, KD, k0, lane);
    v16h b0 = loadB16(Bt, colbase      + hl, KD, k0, lane);
    v16h b1 = loadB16(Bt, colbase + 16 + hl, KD, k0, lane);
    v16h b2 = loadB16(Bt, colbase + 32 + hl, KD, k0, lane);
    v16h b3 = loadB16(Bt, colbase + 48 + hl, KD, k0, lane);
    c0 = WMMA_F16(a, b0, c0);
    c1 = WMMA_F16(a, b1, c1);
    c2 = WMMA_F16(a, b2, c2);
    c3 = WMMA_F16(a, b3, c3);
  }
#pragma unroll
  for (int r = 0; r < 8; ++r) {
    int row = rowbase + up * 8 + r;
    if (row < NODES) {
      float ns = norm_s[row];
      int cb = colbase + hl;
      size_t base = (size_t)row * H1;
      C[base + cb     ] = (c0[r] + bias[cb     ]) * ns;
      C[base + cb + 16] = (c1[r] + bias[cb + 16]) * ns;
      C[base + cb + 32] = (c2[r] + bias[cb + 32]) * ns;
      C[base + cb + 48] = (c3[r] + bias[cb + 48]) * ns;
    }
  }
}

// ---------------- GEMM2: e2 = (h @ W2) * norm_s ; C [NODES,64] ----------------
__global__ void __launch_bounds__(256)
k_gemm2(const _Float16* __restrict__ A, const _Float16* __restrict__ Bt,
        const float* __restrict__ norm_s, float* __restrict__ C) {
  int lane = threadIdx.x & 31, wave = threadIdx.x >> 5;
  int hl = lane & 15, up = lane >> 4;
  int rowbase = blockIdx.x * 128 + wave * 16;
  int arow = rowbase + hl; if (arow >= NODES) arow = NODES - 1;
  v8f c0 = {}, c1 = {}, c2 = {}, c3 = {};
  for (int k0 = 0; k0 < H1; k0 += 32) {
    v16h a  = loadA16(A, arow, H1, k0, lane);
    v16h b0 = loadB16(Bt,      hl, H1, k0, lane);
    v16h b1 = loadB16(Bt, 16 + hl, H1, k0, lane);
    v16h b2 = loadB16(Bt, 32 + hl, H1, k0, lane);
    v16h b3 = loadB16(Bt, 48 + hl, H1, k0, lane);
    c0 = WMMA_F16(a, b0, c0);
    c1 = WMMA_F16(a, b1, c1);
    c2 = WMMA_F16(a, b2, c2);
    c3 = WMMA_F16(a, b3, c3);
  }
#pragma unroll
  for (int r = 0; r < 8; ++r) {
    int row = rowbase + up * 8 + r;
    if (row < NODES) {
      float ns = norm_s[row];
      size_t base = (size_t)row * OUTD;
      C[base + hl     ] = c0[r] * ns;
      C[base + hl + 16] = c1[r] * ns;
      C[base + hl + 32] = c2[r] * ns;
      C[base + hl + 48] = c3[r] * ns;
    }
  }
}

// ---------------- edge scatter-adds ----------------
__global__ void k_scatter256(const int* __restrict__ src, const int* __restrict__ dst,
                             const float* __restrict__ ev, float* __restrict__ agg) {
  int e = blockIdx.x;             // one edge per 256-thread block
  int s = src[e], d = dst[e];
  int c = threadIdx.x;
  atomAddF(&agg[(size_t)d * H1 + c], ev[(size_t)s * H1 + c]);
}
__global__ void k_scatter64(const int* __restrict__ src, const int* __restrict__ dst,
                            const float* __restrict__ ev, float* __restrict__ agg) {
  int e = blockIdx.x * 4 + (threadIdx.x >> 6);  // 4 edges per block
  if (e >= EDGES) return;
  int s = src[e], d = dst[e];
  int c = threadIdx.x & 63;
  atomAddF(&agg[(size_t)d * OUTD + c], ev[(size_t)s * OUTD + c]);
}

// ---------------- layer-1 activation: h = f16(relu(agg*norm_d + b1)) ----------------
__global__ void k_h(const float* __restrict__ agg, const float* __restrict__ norm_d,
                    const float* __restrict__ b1e, _Float16* __restrict__ h) {
  long long idx = (long long)blockIdx.x * blockDim.x + threadIdx.x;
  if (idx >= (long long)NODES * H1) return;
  int n = (int)(idx >> 8), c = (int)(idx & 255);
  float v = agg[idx] * norm_d[n] + b1e[c];
  h[idx] = (_Float16)fmaxf(v, 0.f);
}

// ---------------- gating: fused GEMV + top-2 softmax + importance/load ----------------
__global__ void k_gate(const _Float16* __restrict__ acat, const float* __restrict__ wg,
                       const float* __restrict__ biasg, float* __restrict__ gates,
                       float* imp, float* loadv) {
  int n = blockIdx.x * blockDim.x + threadIdx.x;
  if (n >= NODES) return;
  float g[4] = {biasg[0], biasg[1], biasg[2], biasg[3]};
  const _Float16* a = acat + (size_t)n * KD;
  for (int k = 0; k < KD; ++k) {
    float av = (float)a[k];
    g[0] += av * wg[k * 4 + 0];
    g[1] += av * wg[k * 4 + 1];
    g[2] += av * wg[k * 4 + 2];
    g[3] += av * wg[k * 4 + 3];
  }
  int i0 = 0;
  for (int e = 1; e < 4; ++e) if (g[e] > g[i0]) i0 = e;   // ties -> lowest index
  int i1 = -1;
  for (int e = 0; e < 4; ++e) { if (e == i0) continue; if (i1 < 0 || g[e] > g[i1]) i1 = e; }
  float t = __expf(g[i1] - g[i0]);
  float s = 1.0f / (1.0f + t);
  float g0 = s, g1 = t * s;
  float out[4] = {0.f, 0.f, 0.f, 0.f};
  out[i0] = g0; out[i1] = g1;
  *(float4*)(gates + (size_t)n * 4) = make_float4(out[0], out[1], out[2], out[3]);
  atomAddF(&imp[i0], g0);  atomAddF(&imp[i1], g1);
  atomAddF(&loadv[i0], 1.f); atomAddF(&loadv[i1], 1.f);
}

__global__ void k_loss(const float* __restrict__ imp, const float* __restrict__ loadv,
                       float* __restrict__ out_loss) {
  float mi = 0.f, ml = 0.f;
  for (int e = 0; e < 4; ++e) { mi += imp[e]; ml += loadv[e]; }
  mi *= 0.25f; ml *= 0.25f;
  float vi = 0.f, vl = 0.f;
  for (int e = 0; e < 4; ++e) {
    float di = imp[e] - mi, dl = loadv[e] - ml;
    vi += di * di; vl += dl * dl;
  }
  vi /= 3.f; vl /= 3.f;   // ddof=1
  *out_loss = 0.01f * (vi / (mi * mi + 1e-10f) + vl / (ml * ml + 1e-10f));
}

// ---------------- combine: y += gate[:,e] * (agg2*norm_d + b2) ----------------
__global__ void k_combine(const float* __restrict__ agg2, const float* __restrict__ norm_d,
                          const float* __restrict__ b2e, const float* __restrict__ gates,
                          int e, float* __restrict__ y) {
  long long idx = (long long)blockIdx.x * blockDim.x + threadIdx.x;
  if (idx >= (long long)NODES * OUTD) return;
  int n = (int)(idx >> 6), c = (int)(idx & 63);
  float v = agg2[idx] * norm_d[n] + b2e[c];
  y[idx] += gates[(size_t)n * 4 + e] * v;
}

extern "C" void kernel_launch(void* const* d_in, const int* in_sizes, int n_in,
                              void* d_out, int out_size, void* d_ws, size_t ws_size,
                              hipStream_t stream) {
  const float* feats   = (const float*)d_in[0];
  const float* logits  = (const float*)d_in[1];
  const int*   src     = (const int*)d_in[2];
  const int*   dst     = (const int*)d_in[3];
  const float* Wp_gate = (const float*)d_in[4];
  const float* bp_gate = (const float*)d_in[5];
  const float* w_gate  = (const float*)d_in[6];
  const float* Wp_e    = (const float*)d_in[7];
  const float* bp_e    = (const float*)d_in[8];
  const float* W1      = (const float*)d_in[9];
  const float* b1      = (const float*)d_in[10];
  const float* W2      = (const float*)d_in[11];
  const float* b2      = (const float*)d_in[12];
  float* y = (float*)d_out;

  // ---- workspace carve-up (256B aligned) ----
  uint8_t* base = (uint8_t*)d_ws;
  size_t off = 0;
  auto carve = [&](size_t bytes) { void* p = base + off; off = (off + bytes + 255) & ~(size_t)255; return p; };
  _Float16* acat  = (_Float16*)carve((size_t)NODES * KD * 2);
  float*    e1    = (float*)   carve((size_t)NODES * H1 * 4);
  float*    agg1  = (float*)   carve((size_t)NODES * H1 * 4);
  _Float16* hbuf  = (_Float16*)carve((size_t)NODES * H1 * 2);
  float*    e2    = (float*)   carve((size_t)NODES * OUTD * 4);
  float*    agg2  = (float*)   carve((size_t)NODES * OUTD * 4);
  float*    norm_s= (float*)   carve((size_t)NODES * 4);
  float*    norm_d= (float*)   carve((size_t)NODES * 4);
  float*    gates = (float*)   carve((size_t)NODES * 4 * 4);
  float*    wg    = (float*)   carve((size_t)KD * NE * 4);
  float*    biasg = (float*)   carve(NE * 4);
  _Float16* bt1   = (_Float16*)carve((size_t)NE * H1 * KD * 2);
  float*    b1f   = (float*)   carve((size_t)NE * H1 * 4);
  _Float16* bt2   = (_Float16*)carve((size_t)NE * OUTD * H1 * 2);
  float*    imp   = (float*)   carve(NE * 4);
  float*    loadv = (float*)   carve(NE * 4);
  (void)ws_size; (void)in_sizes; (void)n_in; (void)out_size;

  const int T = 256;
  // degrees -> norms
  hipMemsetAsync(norm_s, 0, (size_t)NODES * 4, stream);
  hipMemsetAsync(norm_d, 0, (size_t)NODES * 4, stream);
  k_degree<<<(EDGES + T - 1) / T, T, 0, stream>>>(src, dst, norm_s, norm_d);
  k_norm<<<(NODES + T - 1) / T, T, 0, stream>>>(norm_s, norm_d);

  // f16 activations + fused weights
  k_acat<<<(int)(((long long)NODES * KD + T - 1) / T), T, 0, stream>>>(feats, logits, acat);
  k_prep_wg<<<(KD * NE + T - 1) / T, T, 0, stream>>>(Wp_gate, w_gate, wg);
  k_prep_biasg<<<1, NE, 0, stream>>>(bp_gate, w_gate, biasg);
  k_prep_bt1<<<(NE * H1 * KD + T - 1) / T, T, 0, stream>>>(Wp_e, W1, bt1);
  k_prep_b1f<<<(NE * H1 + T - 1) / T, T, 0, stream>>>(bp_e, W1, b1f);
  k_prep_bt2<<<(NE * OUTD * H1 + T - 1) / T, T, 0, stream>>>(W2, bt2);

  // gating + loss
  hipMemsetAsync(imp, 0, NE * 4, stream);
  hipMemsetAsync(loadv, 0, NE * 4, stream);
  k_gate<<<(NODES + T - 1) / T, T, 0, stream>>>(acat, wg, biasg, gates, imp, loadv);
  k_loss<<<1, 1, 0, stream>>>(imp, loadv, y + (size_t)NODES * OUTD);

  // experts
  hipMemsetAsync(y, 0, (size_t)NODES * OUTD * 4, stream);
  const int rowBlocks = (NODES + 127) / 128;
  for (int e = 0; e < NE; ++e) {
    k_gemm1<<<dim3(rowBlocks, H1 / 64), T, 0, stream>>>(
        acat, bt1 + (size_t)e * H1 * KD, b1f + (size_t)e * H1, norm_s, e1);
    hipMemsetAsync(agg1, 0, (size_t)NODES * H1 * 4, stream);
    k_scatter256<<<EDGES, T, 0, stream>>>(src, dst, e1, agg1);
    k_h<<<(int)(((long long)NODES * H1 + T - 1) / T), T, 0, stream>>>(
        agg1, norm_d, b1 + (size_t)e * H1, hbuf);
    k_gemm2<<<rowBlocks, T, 0, stream>>>(hbuf, bt2 + (size_t)e * OUTD * H1, norm_s, e2);
    hipMemsetAsync(agg2, 0, (size_t)NODES * OUTD * 4, stream);
    k_scatter64<<<(EDGES + 3) / 4, T, 0, stream>>>(src, dst, e2, agg2);
    k_combine<<<(int)(((long long)NODES * OUTD + T - 1) / T), T, 0, stream>>>(
        agg2, norm_d, b2 + (size_t)e * OUTD, gates, e, y);
  }
}
